// BaselineModel_74354473828636
// MI455X (gfx1250) — matrix operand order repaired and checked by reading, not compiled, not run on hardware
//
#include <hip/hip_runtime.h>
#include <hip/hip_bf16.h>

#define DEV __device__ __forceinline__

typedef __bf16 bf16;
typedef __attribute__((ext_vector_type(16))) __bf16 v16bf;
typedef __attribute__((ext_vector_type(8)))  __bf16 v8bf;
typedef __attribute__((ext_vector_type(8)))  float  v8f;

constexpr int Bc  = 16;
constexpr int TSc = 128;
constexpr int TTc = 128;
constexpr int Hc  = 1024;
constexpr int Vc  = 32000;
constexpr int G4H = 4 * Hc;

DEV bf16  f2bf(float x) { return (bf16)x; }
DEV float sigm(float x) { return 1.0f / (1.0f + __expf(-x)); }

// ---------------------------------------------------------------------------
// Elementwise f32 -> bf16 conversion (grid-stride)
// ---------------------------------------------------------------------------
__global__ void k_f32_to_bf16(const float* __restrict__ src, bf16* __restrict__ dst, size_t n) {
    size_t i = (size_t)blockIdx.x * blockDim.x + threadIdx.x;
    size_t stride = (size_t)gridDim.x * blockDim.x;
    for (; i < n; i += stride) dst[i] = f2bf(src[i]);
}

// ---------------------------------------------------------------------------
// Embedding gather -> bf16 rows  (x[r,:] = emb[tok[r],:])
// ---------------------------------------------------------------------------
__global__ void k_embed_bf16(const float* __restrict__ emb, const int* __restrict__ tok,
                             bf16* __restrict__ dst, int rows) {
    int i = blockIdx.x * blockDim.x + threadIdx.x;
    if (i >= rows * Hc) return;
    int r = i / Hc, j = i - r * Hc;
    dst[i] = f2bf(emb[(size_t)tok[r] * Hc + j]);
}

// ---------------------------------------------------------------------------
// Zero-init LSTM state (c fp32 and h bf16), [B,H]
// ---------------------------------------------------------------------------
__global__ void k_zero_state(float* __restrict__ c, bf16* __restrict__ h) {
    int i = blockIdx.x * blockDim.x + threadIdx.x;
    if (i >= Bc * Hc) return;
    c[i] = 0.0f;
    h[i] = (bf16)0.0f;
}

// ---------------------------------------------------------------------------
// Tiled bf16 WMMA GEMM:  C[M,N] (+bias) = A[M,K] @ W[N,K]^T
// One wave computes a 16x64 tile: A-fragment reused across 4 v_wmma per k-step.
// All dims assumed multiples of tile sizes (true for this model).
// ---------------------------------------------------------------------------
__global__ void __launch_bounds__(256) k_gemm_bf16(
    const bf16* __restrict__ A, const bf16* __restrict__ W,
    const float* __restrict__ bias, float* __restrict__ C,
    int M, int N, int K)
{
    const int lane = threadIdx.x & 31;
    const int wave = blockIdx.x * (blockDim.x >> 5) + (threadIdx.x >> 5);
    const int nT   = N >> 6;                    // number of 64-wide column tiles
    const int total = (M >> 4) * nT;
    if (wave >= total) return;                  // wave-uniform exit (EXEC all-ones inside)
    const int mt = wave / nT;
    const int nt = wave - mt * nT;

    const int lmod = lane & 15;
    const int kh   = lane >> 4;                 // half-wave selector

    // A: lane holds row (mt*16 + lane%16); K-chunks {kh*8 .. +7} and {16+kh*8 .. +7}
    const bf16* Arow = A + (size_t)(mt * 16 + lmod) * K;
    // B: lane holds column (nt*64 + j*16 + lane%16); K-chunks {kh*8 .. +7} and {16+kh*8 .. +7}
    const bf16* W0   = W + (size_t)(nt * 64 + lmod) * K;

    v8f acc[4] = {};
    for (int k = 0; k < K; k += 32) {
        union { v16bf v; v8bf h[2]; } a;
        a.h[0] = *(const v8bf*)(Arow + k +      kh * 8);
        a.h[1] = *(const v8bf*)(Arow + k + 16 + kh * 8);
#pragma unroll
        for (int j = 0; j < 4; ++j) {
            union { v16bf v; v8bf h[2]; } bfrag;
            bfrag.h[0] = *(const v8bf*)(W0 + (size_t)j * 16 * K + k +      kh * 8);
            bfrag.h[1] = *(const v8bf*)(W0 + (size_t)j * 16 * K + k + 16 + kh * 8);
            acc[j] = __builtin_amdgcn_wmma_f32_16x16x32_bf16(
                false, a.v, false, bfrag.v, (short)0, acc[j], false, false);
        }
    }

    // D layout: VGPR r, lane L -> row = r + 8*(L/16), col = L%16 (within 16-col tile)
    const int rbase = mt * 16 + kh * 8;
#pragma unroll
    for (int j = 0; j < 4; ++j) {
        const int ncol = nt * 64 + j * 16 + lmod;
        const float bv = bias ? bias[ncol] : 0.0f;
#pragma unroll
        for (int r = 0; r < 8; ++r)
            C[(size_t)(rbase + r) * N + ncol] = acc[j][r] + bv;
    }
}

// ---------------------------------------------------------------------------
// LSTM pointwise step: gates = xw[b,t,:] + hw[b,:] + bih + bhh  (i,f,g,o)
// c' = sig(f)*c + sig(i)*tanh(g);  h' = sig(o)*tanh(c')
// Writes h' fp32 into out[b,t,:] and bf16 into h_bf for the next step's GEMM.
// ---------------------------------------------------------------------------
__global__ void k_lstm_pointwise(
    const float* __restrict__ xw,   // [B*T, 4H] precomputed x@Wih^T
    const float* __restrict__ hw,   // [B, 4H]   this step's h@Whh^T
    const float* __restrict__ bih, const float* __restrict__ bhh,
    float* __restrict__ cstate,     // [B, H]
    float* __restrict__ out,        // [B*T, H]
    bf16*  __restrict__ h_bf,       // [B, H]
    int t, int T)
{
    int idx = blockIdx.x * blockDim.x + threadIdx.x;
    if (idx >= Bc * Hc) return;
    int b = idx / Hc, j = idx - b * Hc;
    const float* g  = xw + (size_t)(b * T + t) * G4H;
    const float* gh = hw + (size_t)b * G4H;

    float gi = g[j]          + gh[j]          + bih[j]          + bhh[j];
    float gf = g[Hc + j]     + gh[Hc + j]     + bih[Hc + j]     + bhh[Hc + j];
    float gg = g[2 * Hc + j] + gh[2 * Hc + j] + bih[2 * Hc + j] + bhh[2 * Hc + j];
    float go = g[3 * Hc + j] + gh[3 * Hc + j] + bih[3 * Hc + j] + bhh[3 * Hc + j];

    float c = sigm(gf) * cstate[idx] + sigm(gi) * tanhf(gg);
    float h = sigm(go) * tanhf(c);
    cstate[idx] = c;
    out[(size_t)(b * T + t) * Hc + j] = h;
    h_bf[idx] = f2bf(h);
}

// ---------------------------------------------------------------------------
// Attention scores along the softmax axis: se[b,e] = enc_out[b,e,:] . we
// (dec.wd and attn_b are constant along axis e -> drop out of the softmax)
// One wave per (b,e) dot product, wave32 shuffle reduction.
// ---------------------------------------------------------------------------
__global__ void k_attn_scores(const float* __restrict__ enc_out,
                              const float* __restrict__ attn_w,
                              float* __restrict__ se) {
    int wave = (blockIdx.x * blockDim.x + threadIdx.x) >> 5;
    int lane = threadIdx.x & 31;
    if (wave >= Bc * TSc) return;
    const float* row = enc_out + (size_t)wave * Hc;
    const float* we  = attn_w + Hc;
    float s = 0.0f;
    for (int j = lane; j < Hc; j += 32) s += row[j] * we[j];
#pragma unroll
    for (int off = 16; off > 0; off >>= 1) s += __shfl_xor(s, off, 32);
    if (lane == 0) se[wave] = s;
}

// ---------------------------------------------------------------------------
// Per-batch softmax over TS + weighted sum of encoder states -> ctx[b,:]
// ---------------------------------------------------------------------------
__global__ void k_softmax_ctx(const float* __restrict__ se,
                              const float* __restrict__ enc_out,
                              float* __restrict__ ctx) {
    __shared__ float p[TSc];
    const int b = blockIdx.x;
    if (threadIdx.x == 0) {
        float mx = -3.4e38f;
        for (int e = 0; e < TSc; ++e) mx = fmaxf(mx, se[b * TSc + e]);
        float sum = 0.0f;
        for (int e = 0; e < TSc; ++e) { p[e] = __expf(se[b * TSc + e] - mx); sum += p[e]; }
        float inv = 1.0f / sum;
        for (int e = 0; e < TSc; ++e) p[e] *= inv;
    }
    __syncthreads();
    for (int j = threadIdx.x; j < Hc; j += blockDim.x) {
        float acc = 0.0f;
        for (int e = 0; e < TSc; ++e)
            acc += p[e] * enc_out[((size_t)b * TSc + e) * Hc + j];
        ctx[b * Hc + j] = acc;
    }
}

// ---------------------------------------------------------------------------
// Build bf16 concat [dec_out | ctx(broadcast over t)] : [B*TT, 2H]
// ---------------------------------------------------------------------------
__global__ void k_build_cat(const float* __restrict__ dec_out,
                            const float* __restrict__ ctx,
                            bf16* __restrict__ cat) {
    int i = blockIdx.x * blockDim.x + threadIdx.x;
    if (i >= Bc * TTc * Hc) return;
    int bt = i / Hc, j = i - bt * Hc;
    int b = bt / TTc;
    cat[(size_t)bt * (2 * Hc) + j]      = f2bf(dec_out[i]);
    cat[(size_t)bt * (2 * Hc) + Hc + j] = f2bf(ctx[b * Hc + j]);
}

// ---------------------------------------------------------------------------
extern "C" void kernel_launch(void* const* d_in, const int* in_sizes, int n_in,
                              void* d_out, int out_size, void* d_ws, size_t ws_size,
                              hipStream_t stream)
{
    const int*   src     = (const int*)  d_in[0];
    const int*   tgt     = (const int*)  d_in[1];
    const float* src_emb = (const float*)d_in[2];
    const float* tgt_emb = (const float*)d_in[3];
    const float* enc_Wih = (const float*)d_in[4];
    const float* enc_Whh = (const float*)d_in[5];
    const float* enc_bih = (const float*)d_in[6];
    const float* enc_bhh = (const float*)d_in[7];
    const float* dec_Wih = (const float*)d_in[8];
    const float* dec_Whh = (const float*)d_in[9];
    const float* dec_bih = (const float*)d_in[10];
    const float* dec_bhh = (const float*)d_in[11];
    const float* attn_w  = (const float*)d_in[12];
    // d_in[13] = attn_b : constant along softmax axis -> mathematically drops out
    const float* fc_w    = (const float*)d_in[14];
    const float* fc_b    = (const float*)d_in[15];
    float* out = (float*)d_out;

    // ---- workspace bump allocator (deterministic layout every call) ----
    char* p = (char*)d_ws;
    auto alloc = [&](size_t bytes) -> void* {
        void* r = (void*)p;
        p += (bytes + 255) & ~(size_t)255;
        return r;
    };

    bf16*  wih_e  = (bf16*) alloc((size_t)G4H * Hc * sizeof(bf16));      // enc_Wih bf16
    bf16*  whh_e  = (bf16*) alloc((size_t)G4H * Hc * sizeof(bf16));      // enc_Whh bf16
    bf16*  wih_d  = (bf16*) alloc((size_t)G4H * Hc * sizeof(bf16));      // dec_Wih bf16
    bf16*  whh_d  = (bf16*) alloc((size_t)G4H * Hc * sizeof(bf16));      // dec_Whh bf16
    bf16*  fcw_bf = (bf16*) alloc((size_t)Vc * 2 * Hc * sizeof(bf16));   // fc_w bf16
    bf16*  x_bf   = (bf16*) alloc((size_t)Bc * TSc * Hc * sizeof(bf16)); // embedded inputs (reused)
    float* xw_enc = (float*)alloc((size_t)Bc * TSc * G4H * sizeof(float));
    float* xw_dec = (float*)alloc((size_t)Bc * TTc * G4H * sizeof(float));
    float* gates  = (float*)alloc((size_t)Bc * G4H * sizeof(float));     // per-step h@Whh^T
    float* cstate = (float*)alloc((size_t)Bc * Hc * sizeof(float));
    bf16*  h_bf   = (bf16*) alloc((size_t)Bc * Hc * sizeof(bf16));
    float* enc_o  = (float*)alloc((size_t)Bc * TSc * Hc * sizeof(float));
    float* dec_o  = (float*)alloc((size_t)Bc * TTc * Hc * sizeof(float));
    float* se     = (float*)alloc((size_t)Bc * TSc * sizeof(float));
    float* ctx    = (float*)alloc((size_t)Bc * Hc * sizeof(float));
    bf16*  catb   = (bf16*) alloc((size_t)Bc * TTc * 2 * Hc * sizeof(bf16));
    (void)ws_size; (void)in_sizes; (void)n_in; (void)out_size;

    // ---- 1) weight down-conversions to bf16 (fc_w stays hot in L2 after this) ----
    k_f32_to_bf16<<<4096, 256, 0, stream>>>(enc_Wih, wih_e, (size_t)G4H * Hc);
    k_f32_to_bf16<<<4096, 256, 0, stream>>>(enc_Whh, whh_e, (size_t)G4H * Hc);
    k_f32_to_bf16<<<4096, 256, 0, stream>>>(dec_Wih, wih_d, (size_t)G4H * Hc);
    k_f32_to_bf16<<<4096, 256, 0, stream>>>(dec_Whh, whh_d, (size_t)G4H * Hc);
    k_f32_to_bf16<<<8192, 256, 0, stream>>>(fc_w, fcw_bf, (size_t)Vc * 2 * Hc);

    // ---- 2) init LSTM state ----
    k_zero_state<<<(Bc * Hc) / 256, 256, 0, stream>>>(cstate, h_bf);

    // ---- 3) hoisted input projections: x@Wih^T for the whole sequence ----
    // encoder: [2048,1024]@[1024,4096]
    k_embed_bf16<<<(Bc * TSc * Hc) / 256, 256, 0, stream>>>(src_emb, src, x_bf, Bc * TSc);
    k_gemm_bf16<<<(((Bc * TSc) / 16) * (G4H / 64)) / 8, 256, 0, stream>>>(
        x_bf, wih_e, nullptr, xw_enc, Bc * TSc, G4H, Hc);
    // decoder (x_bf reused; stream-ordered after encoder GEMM)
    k_embed_bf16<<<(Bc * TTc * Hc) / 256, 256, 0, stream>>>(tgt_emb, tgt, x_bf, Bc * TTc);
    k_gemm_bf16<<<(((Bc * TTc) / 16) * (G4H / 64)) / 8, 256, 0, stream>>>(
        x_bf, wih_d, nullptr, xw_dec, Bc * TTc, G4H, Hc);

    // ---- 4) encoder recurrence: per step, [16,1024]@[1024,4096] + pointwise ----
    const int recWaves  = (16 / 16) * (G4H / 64);      // 64 waves
    const int recBlocks = recWaves / 8;                // 8 blocks of 256 threads
    for (int t = 0; t < TSc; ++t) {
        k_gemm_bf16<<<recBlocks, 256, 0, stream>>>(h_bf, whh_e, nullptr, gates, 16, G4H, Hc);
        k_lstm_pointwise<<<(Bc * Hc) / 256, 256, 0, stream>>>(
            xw_enc, gates, enc_bih, enc_bhh, cstate, enc_o, h_bf, t, TSc);
    }

    // ---- 5) decoder recurrence (h/c carry over from encoder = (hT, cT)) ----
    for (int t = 0; t < TTc; ++t) {
        k_gemm_bf16<<<recBlocks, 256, 0, stream>>>(h_bf, whh_d, nullptr, gates, 16, G4H, Hc);
        k_lstm_pointwise<<<(Bc * Hc) / 256, 256, 0, stream>>>(
            xw_dec, gates, dec_bih, dec_bhh, cstate, dec_o, h_bf, t, TTc);
    }

    // ---- 6) attention (softmax is shift-invariant -> t-independent context) ----
    k_attn_scores<<<(Bc * TSc * 32) / 256, 256, 0, stream>>>(enc_o, attn_w, se);
    k_softmax_ctx<<<Bc, 256, 0, stream>>>(se, enc_o, ctx);

    // ---- 7) concat + final FC: [2048,2048]@[2048,32000] + bias -> d_out ----
    k_build_cat<<<(Bc * TTc * Hc) / 256, 256, 0, stream>>>(dec_o, ctx, catb);
    k_gemm_bf16<<<(((Bc * TTc) / 16) * (Vc / 64)) / 8, 256, 0, stream>>>(
        catb, fcw_bf, fc_b, out, Bc * TTc, Vc, 2 * Hc);
}
